// Net_12610023981837
// MI455X (gfx1250) — matrix operand order repaired and checked by reading, not compiled, not run on hardware
//
#include <hip/hip_runtime.h>
#include <hip/hip_bf16.h>
#include <math.h>

typedef _Float16 h16;
typedef __attribute__((ext_vector_type(16))) _Float16 v16h;
typedef __attribute__((ext_vector_type(8)))  _Float16 v8h;
typedef __attribute__((ext_vector_type(8)))  float    v8f;

#define NB    256      // batch
#define DIM   48
#define NC    10
#define D2    2304     // DIM*DIM
#define K2    288      // conv2 K = 9 taps * 32 channels (k = r*32 + ci)
#define FC1K  12544
#define NBP   576      // class-sorted sample columns, padded (<= 256 + 10*31, mult of 32)

__device__ __forceinline__ v8f wmma16(v16h a, v16h b, v8f c) {
  return __builtin_amdgcn_wmma_f32_16x16x32_f16(false, a, false, b, (short)0, c, false, false);
}
__device__ __forceinline__ v16h cat8(v8h lo, v8h hi) {
  return __builtin_shufflevector(lo, hi, 0, 1, 2, 3, 4, 5, 6, 7, 8, 9, 10, 11, 12, 13, 14, 15);
}

// signed_root: sign(sign(m)+0.1) * ((|m|+a)^p - b);  +1 for m>=0, -1 for m<0
__device__ __forceinline__ float sr2(float m) {
  float s = (m < 0.f) ? -1.f : 1.f;
  return s * (sqrtf(fabsf(m) + 0.25f) - 0.5f);
}
__device__ __forceinline__ float sr3(float m) {
  float s = (m < 0.f) ? -1.f : 1.f;
  return s * (cbrtf(fabsf(m) + 0.19245008973f) - 0.57735026919f);
}
__device__ __forceinline__ float sr4(float m) {
  float s = (m < 0.f) ? -1.f : 1.f;
  return s * (sqrtf(sqrtf(fabsf(m) + 0.15749013123f)) - 0.62996052494f);
}

// ---------------------------------------------------------------- init / cast
__global__ void k_init(float* p) {
  if (threadIdx.x < 4) p[threadIdx.x] = 0.f;
}

__global__ void k_cast(const float* __restrict__ src, h16* __restrict__ dst, int n) {
  int i = blockIdx.x * blockDim.x + threadIdx.x;
  if (i < n) dst[i] = (h16)src[i];
}

// conv2 weights [64,32,3,3] -> w2k[oc][r*32+ci] f16 (tap-major K, matches im2col chunks)
__global__ void k_reord2(const float* __restrict__ w, h16* __restrict__ w2k) {
  int tid = blockIdx.x * blockDim.x + threadIdx.x;           // < 64*288
  int oc = tid / K2, k = tid % K2, r = k >> 5, ci = k & 31;
  w2k[tid] = (h16)w[(oc * 32 + ci) * 9 + r];
}

// ---------------------------------------------------------------- conv1 (3ch, direct f32)
// out pixel-major: h1p[b][pix(30*30)][oc(32)]; lanes span oc -> broadcast x loads, coalesced stores
__global__ void k_conv1(const float* __restrict__ x, const float* __restrict__ w,
                        const float* __restrict__ bias, h16* __restrict__ out) {
  int tid = blockIdx.x * blockDim.x + threadIdx.x;           // < 256*900*32
  int oc = tid & 31, pix = (tid >> 5) % 900, b = tid / 28800;
  int oy = pix / 30, ox = pix % 30;
  float s = bias[oc];
  #pragma unroll
  for (int ci = 0; ci < 3; ++ci)
    #pragma unroll
    for (int dy = 0; dy < 3; ++dy)
      #pragma unroll
      for (int dx = 0; dx < 3; ++dx)
        s += x[((b * 3 + ci) * 32 + oy + dy) * 32 + (ox + dx)] *
             w[((oc * 3 + ci) * 3 + dy) * 3 + dx];
  out[tid] = (h16)fmaxf(s, 0.f);
}

// ---------------------------------------------------------------- conv2 implicit-GEMM WMMA
// h1p:[B,900,32] f16 staged in LDS; w2k:[64,288] f16 (tap-major K); out:[B,64,784] (bias+relu)
__global__ void k_conv2(const h16* __restrict__ h1p, const h16* __restrict__ w2k,
                        const float* __restrict__ b2, h16* __restrict__ out) {
  __shared__ h16 s_h1[900 * 32];                             // 57.6 KB of 320 KB WGP LDS
  int b = blockIdx.x;
  int lane = threadIdx.x & 31, wave = threadIdx.x >> 5;      // 8 waves / block
  {
    const v8h* src = (const v8h*)(h1p + (size_t)b * 900 * 32);
    v8h* dst = (v8h*)s_h1;
    for (int i = threadIdx.x; i < 3600; i += 256) dst[i] = src[i];
  }
  __syncthreads();
  for (int t = wave; t < 4 * 49; t += 8) {                   // 4 m-tiles x 49 n-tiles
    int mt = t / 49, nt = t % 49;
    v8f acc = {};
    int m   = mt * 16 + (lane & 15);
    int pix = nt * 16 + (lane & 15);
    int oy  = pix / 28, ox = pix % 28;
    const h16* arow = w2k + m * K2 + ((lane >> 4) << 4);
    const h16* srow = s_h1 + ((lane >> 4) << 4);
    #pragma unroll
    for (int kc = 0; kc < 9; ++kc) {                         // one filter tap per 32-wide chunk
      const int dy = kc / 3, dx = kc % 3;                    // constants after unroll
      const v8h* pa = (const v8h*)(arow + kc * 32);
      v16h af = cat8(pa[0], pa[1]);
      const v8h* pb = (const v8h*)(srow + ((oy + dy) * 30 + ox + dx) * 32);
      v16h bf = cat8(pb[0], pb[1]);
      acc = wmma16(af, bf, acc);
    }
    #pragma unroll
    for (int r = 0; r < 8; ++r) {
      int mm = mt * 16 + r + ((lane >> 4) << 3);
      int nn = nt * 16 + (lane & 15);
      out[((size_t)b * 64 + mm) * 784 + nn] = (h16)fmaxf(acc[r] + b2[mm], 0.f);
    }
  }
}

// ---------------------------------------------------------------- 2x2 maxpool + flatten
__global__ void k_pool(const h16* __restrict__ in, h16* __restrict__ out) {
  int tid = blockIdx.x * blockDim.x + threadIdx.x;           // < 256*12544
  int px = tid % 14, py = (tid / 14) % 14, ch = (tid / 196) % 64, b = tid / FC1K;
  const h16* p = in + ((size_t)b * 64 + ch) * 784 + (2 * py) * 28 + 2 * px;
  float v = fmaxf(fmaxf((float)p[0], (float)p[1]), fmaxf((float)p[28], (float)p[29]));
  out[tid] = (h16)v;
}

// ---------------------------------------------------------------- generic WMMA GEMM
// out[m,n] = relu?( A[m,:K] . W[n,:K] + bias[n] );  A:[M,lda] f16, W:[N,ldw] f16
__global__ void k_gemm(const h16* __restrict__ A, int lda, const h16* __restrict__ W, int ldw,
                       const float* __restrict__ bias, h16* outH, float* outF,
                       int M, int N, int K, int relu) {
  int lane = threadIdx.x & 31, wave = threadIdx.x >> 5;
  int mtiles = M >> 4, ntiles = N >> 4, kch = K >> 5;
  int wpb = blockDim.x >> 5;
  for (int t = blockIdx.x * wpb + wave; t < mtiles * ntiles; t += gridDim.x * wpb) {
    int mt = t / ntiles, nt = t % ntiles;
    int m = mt * 16 + (lane & 15);
    int n = nt * 16 + (lane & 15);
    const h16* arow = A + (size_t)m * lda + ((lane >> 4) << 4);
    const h16* brow = W + (size_t)n * ldw + ((lane >> 4) << 4);
    v8f acc = {};
    for (int kc = 0; kc < kch; ++kc) {
      const v8h* pa = (const v8h*)(arow + (kc << 5));
      const v8h* pb = (const v8h*)(brow + (kc << 5));
      __builtin_prefetch(&pa[8], 0, 1);
      __builtin_prefetch(&pb[8], 0, 1);
      v16h af = cat8(pa[0], pa[1]);
      v16h bf = cat8(pb[0], pb[1]);
      acc = wmma16(af, bf, acc);
    }
    #pragma unroll
    for (int r = 0; r < 8; ++r) {
      int mm = mt * 16 + r + ((lane >> 4) << 3);
      int nn = nt * 16 + (lane & 15);
      float v = acc[r] + bias[nn];
      if (relu) v = fmaxf(v, 0.f);
      if (outH) outH[(size_t)mm * N + nn] = (h16)v;
      else      outF[(size_t)mm * N + nn] = v;
    }
  }
}

// ---------------------------------------------------------------- GMM head: log_resp + do
__global__ void k_head(const float* __restrict__ z, const float* __restrict__ centers,
                       float* __restrict__ log_resp, float* __restrict__ dof,
                       int* __restrict__ cls) {
  int b = blockIdx.x * blockDim.x + threadIdx.x;
  if (b >= NB) return;
  float zv[DIM];
  #pragma unroll
  for (int d = 0; d < DIM; ++d) zv[d] = z[b * DIM + d];
  float dist[NC], dmin = 3.4e38f;
  int amin = 0;
  for (int c = 0; c < NC; ++c) {
    float s = 0.f;
    for (int d = 0; d < DIM; ++d) {
      float t = zv[d] - centers[c * DIM + d];
      s += t * t;
    }
    dist[c] = s;
    if (s < dmin) { dmin = s; amin = c; }                    // first-min == jnp.argmin
  }
  float mx = -0.5f * dmin;                                   // max of expo
  float ssum = 0.f;
  for (int c = 0; c < NC; ++c) ssum += expf(-0.5f * dist[c] - mx);
  for (int c = 0; c < NC; ++c) {
    float r = expf(-0.5f * dist[c] - mx) / ssum;
    log_resp[b * NC + c] = logf(fmaxf(r, 1e-8f));
  }
  cls[b] = amin;
  for (int d = 0; d < DIM; ++d)
    dof[b * DIM + d] = zv[d] - centers[amin * DIM + d];
}

// class-sorted sample map: idx lists, counts, weights, padded class offsets, srcmap[NBP]
__global__ void k_lists(const int* __restrict__ cls, int* __restrict__ idx,
                        int* __restrict__ cnt, float* __restrict__ cwn,
                        int* __restrict__ off, int* __restrict__ srcmap) {
  if (threadIdx.x != 0) return;
  int o = 0;
  for (int c = 0; c < NC; ++c) {
    int n = 0;
    for (int b = 0; b < NB; ++b)
      if (cls[b] == c) idx[c * NB + n++] = b;
    cnt[c] = n;
    cwn[c] = (float)n / (float)NB;                           // sum(cw) == B always
    off[c] = o;
    for (int s = 0; s < n; ++s) srcmap[o + s] = idx[c * NB + s];
    int pad = ((n + 31) / 32) * 32;
    for (int s = n; s < pad; ++s) srcmap[o + s] = -1;
    o += pad;
  }
  for (; o < NBP; ++o) srcmap[o] = -1;
}

// build class-sorted transposed layouts:
// pairT[p][col] = do[b,i]*do[b,j], doT[k][col] = do[b,k], zero for pad columns
__global__ void k_build(const float* __restrict__ dof, const int* __restrict__ srcmap,
                        h16* __restrict__ pairT, h16* __restrict__ doT) {
  int tid = blockIdx.x * blockDim.x + threadIdx.x;           // < D2*NBP = 1,327,104
  int col = tid % NBP, p = tid / NBP;
  int b = srcmap[col];
  float v = (b >= 0) ? dof[b * DIM + p / DIM] * dof[b * DIM + p % DIM] : 0.f;
  pairT[tid] = (h16)v;
  if (p < DIM) doT[p * NBP + col] = (h16)((b >= 0) ? dof[b * DIM + p] : 0.f);
}

// ---------------------------------------------------------------- p1, p2 (streaming)
__global__ void k_p1(const float* __restrict__ dof, const int* __restrict__ idx,
                     const int* __restrict__ cnt, const float* __restrict__ cwn,
                     const float* __restrict__ gm1, const float* __restrict__ mw,
                     float* __restrict__ out) {
  int tid = blockIdx.x * blockDim.x + threadIdx.x;           // c*48+i
  if (tid >= NC * DIM) return;
  int c = tid / DIM, i = tid % DIM;
  float s = 0.f;
  int n = cnt[c];
  for (int k = 0; k < n; ++k) s += dof[idx[c * NB + k] * DIM + i];
  float m1 = s / ((float)n + 1e-7f);
  float v = m1 - gm1[i];                                     // t1 = gm1 (no signed root)
  atomicAdd(out, cwn[c] * mw[i] * v * v);
}

__global__ void k_p2(const float* __restrict__ dof, const int* __restrict__ idx,
                     const int* __restrict__ cnt, const float* __restrict__ cwn,
                     const float* __restrict__ gm2, const float* __restrict__ mw,
                     float* __restrict__ out) {
  int tid = blockIdx.x * blockDim.x + threadIdx.x;           // c*2304+p
  if (tid >= NC * D2) return;
  int c = tid / D2, p = tid % D2, i = p / DIM, j = p % DIM;
  float s = 0.f;
  int n = cnt[c];
  for (int k = 0; k < n; ++k) {
    const float* d = dof + idx[c * NB + k] * DIM;
    s += d[i] * d[j];
  }
  float m2 = s / ((float)n + 1e-7f);
  float v = sr2(m2) - sr2(gm2[p]);
  atomicAdd(out, cwn[c] * mw[j] * v * v);
}

// ---------------------------------------------------------------- p3 via WMMA grams
// m3[c,p,k] = sum_{b in class c} pair[b,p]*do[b,k]   (NOT count-normalized)
__global__ void k_p3(const h16* __restrict__ pairT, const h16* __restrict__ doT,
                     const int* __restrict__ cnt, const float* __restrict__ cwn,
                     const int* __restrict__ off, const float* __restrict__ gm3,
                     const float* __restrict__ mw, float* __restrict__ out) {
  int lane = threadIdx.x & 31, wave = threadIdx.x >> 5;
  int t = blockIdx.x * (blockDim.x >> 5) + wave;             // 144 p-tiles x 3 n-tiles
  if (t >= 144 * 3) return;
  int pt = t / 3, nt = t % 3;
  int n = nt * 16 + (lane & 15);                             // k-axis index (last axis)
  float mwn = mw[n];
  v8f t3f;
  #pragma unroll
  for (int r = 0; r < 8; ++r) {
    int pp = pt * 16 + r + ((lane >> 4) << 3);
    t3f[r] = sr3(gm3[(size_t)pp * DIM + n]);
  }
  int m = pt * 16 + (lane & 15);
  const h16* arow = pairT + (size_t)m * NBP + ((lane >> 4) << 4);
  const h16* brow = doT + (size_t)n * NBP + ((lane >> 4) << 4);
  float local = 0.f;
  for (int c = 0; c < NC; ++c) {
    v8f acc = {};
    int base = off[c], nch = (cnt[c] + 31) >> 5;
    for (int kc = 0; kc < nch; ++kc) {
      const v8h* pa = (const v8h*)(arow + base + kc * 32);
      const v8h* pb = (const v8h*)(brow + base + kc * 32);
      v16h af = cat8(pa[0], pa[1]);
      v16h bf = cat8(pb[0], pb[1]);
      acc = wmma16(af, bf, acc);
    }
    float cw = cwn[c];
    #pragma unroll
    for (int r = 0; r < 8; ++r) {
      float v = sr3(acc[r]) - t3f[r];
      local += cw * mwn * v * v;
    }
  }
  #pragma unroll
  for (int o = 16; o; o >>= 1) local += __shfl_xor(local, o, 32);
  if (lane == 0) atomicAdd(out, local);
}

// ---------------------------------------------------------------- p4 via WMMA grams, m4 never materialized
// m4[c,p,q] = sum_{b in class c} pair[b,p]*pair[b,q]  (NOT count-normalized)
__global__ void k_p4(const h16* __restrict__ pairT, const int* __restrict__ cnt,
                     const float* __restrict__ cwn, const int* __restrict__ off,
                     const float* __restrict__ gm4, const float* __restrict__ mw,
                     float* __restrict__ out) {
  int lane = threadIdx.x & 31, wave = threadIdx.x >> 5;
  int t = blockIdx.x * (blockDim.x >> 5) + wave;             // 144x144 = 20736 tiles
  if (t >= 144 * 144) return;
  int pt = t / 144, qt = t % 144;
  int q = qt * 16 + (lane & 15);
  float mwn = mw[q % DIM];                                   // mom_weight on LAST axis
  v8f t4f;
  #pragma unroll
  for (int r = 0; r < 8; ++r) {
    int pp = pt * 16 + r + ((lane >> 4) << 3);
    t4f[r] = sr4(gm4[(size_t)pp * D2 + q]);
  }
  int m = pt * 16 + (lane & 15);
  const h16* arow = pairT + (size_t)m * NBP + ((lane >> 4) << 4);
  const h16* brow = pairT + (size_t)q * NBP + ((lane >> 4) << 4);
  float local = 0.f;
  for (int c = 0; c < NC; ++c) {
    v8f acc = {};
    int base = off[c], nch = (cnt[c] + 31) >> 5;
    for (int kc = 0; kc < nch; ++kc) {
      const v8h* pa = (const v8h*)(arow + base + kc * 32);
      const v8h* pb = (const v8h*)(brow + base + kc * 32);
      v16h af = cat8(pa[0], pa[1]);
      v16h bf = cat8(pb[0], pb[1]);
      acc = wmma16(af, bf, acc);
    }
    float cw = cwn[c];
    #pragma unroll
    for (int r = 0; r < 8; ++r) {
      float v = sr4(acc[r]) - t4f[r];
      local += cw * mwn * v * v;
    }
  }
  #pragma unroll
  for (int o = 16; o; o >>= 1) local += __shfl_xor(local, o, 32);
  if (lane == 0) atomicAdd(out, local);
}

// ================================================================ launch
extern "C" void kernel_launch(void* const* d_in, const int* in_sizes, int n_in,
                              void* d_out, int out_size, void* d_ws, size_t ws_size,
                              hipStream_t stream) {
  const float* x    = (const float*)d_in[0];
  const float* c1w  = (const float*)d_in[1];
  const float* c1b  = (const float*)d_in[2];
  const float* c2w  = (const float*)d_in[3];
  const float* c2b  = (const float*)d_in[4];
  const float* f1w  = (const float*)d_in[5];
  const float* f1b  = (const float*)d_in[6];
  const float* f2w  = (const float*)d_in[7];
  const float* f2b  = (const float*)d_in[8];
  const float* f3w  = (const float*)d_in[9];
  const float* f3b  = (const float*)d_in[10];
  const float* cent = (const float*)d_in[11];
  const float* mw   = (const float*)d_in[12];
  const float* gm1  = (const float*)d_in[13];
  const float* gm2  = (const float*)d_in[14];
  const float* gm3  = (const float*)d_in[15];
  const float* gm4  = (const float*)d_in[16];
  float* outp = (float*)d_out;                               // [0..2559]=log_resp, 2560..3 = p1..p4

  char* base = (char*)d_ws;
  size_t woff = 0;
  auto alloc = [&](size_t bytes) -> void* {
    woff = (woff + 255) & ~(size_t)255;
    void* p = base + woff;
    woff += bytes;
    return p;
  };
  h16*  h1p   = (h16*)alloc((size_t)NB * 900 * 32 * 2);
  h16*  c2out = (h16*)alloc((size_t)NB * 64 * 784 * 2);
  h16*  h2    = (h16*)alloc((size_t)NB * FC1K * 2);
  h16*  w2k   = (h16*)alloc((size_t)64 * K2 * 2);
  h16*  f1wh  = (h16*)alloc((size_t)128 * FC1K * 2);
  h16*  f2wh  = (h16*)alloc((size_t)128 * 128 * 2);
  h16*  f3wh  = (h16*)alloc((size_t)DIM * 128 * 2);
  h16*  h3    = (h16*)alloc((size_t)NB * 128 * 2);
  h16*  h4    = (h16*)alloc((size_t)NB * 128 * 2);
  float* z    = (float*)alloc((size_t)NB * DIM * 4);
  float* dof  = (float*)alloc((size_t)NB * DIM * 4);
  int*  cls   = (int*)alloc((size_t)NB * 4);
  int*  idx   = (int*)alloc((size_t)NC * NB * 4);
  int*  cnt   = (int*)alloc((size_t)NC * 4);
  float* cwn  = (float*)alloc((size_t)NC * 4);
  int*  off   = (int*)alloc((size_t)NC * 4);
  int*  smap  = (int*)alloc((size_t)NBP * 4);
  h16*  pairT = (h16*)alloc((size_t)D2 * NBP * 2);
  h16*  doT   = (h16*)alloc((size_t)DIM * NBP * 2);
  (void)ws_size; (void)n_in; (void)in_sizes; (void)out_size;

  // zero the p1..p4 accumulators every call (graph-replay safe)
  k_init<<<1, 32, 0, stream>>>(outp + 2560);

  // weight reorder / casts to f16
  k_reord2<<<(64 * K2 + 255) / 256, 256, 0, stream>>>(c2w, w2k);
  k_cast<<<(128 * FC1K + 255) / 256, 256, 0, stream>>>(f1w, f1wh, 128 * FC1K);
  k_cast<<<(128 * 128 + 255) / 256, 256, 0, stream>>>(f2w, f2wh, 128 * 128);
  k_cast<<<(DIM * 128 + 255) / 256, 256, 0, stream>>>(f3w, f3wh, DIM * 128);

  // backbone
  k_conv1<<<(NB * 900 * 32) / 256, 256, 0, stream>>>(x, c1w, c1b, h1p);
  k_conv2<<<NB, 256, 0, stream>>>(h1p, w2k, c2b, c2out);
  k_pool<<<(NB * FC1K) / 256, 256, 0, stream>>>(c2out, h2);
  k_gemm<<<32, 128, 0, stream>>>(h2, FC1K, f1wh, FC1K, f1b, h3, nullptr, NB, 128, FC1K, 1);
  k_gemm<<<32, 128, 0, stream>>>(h3, 128, f2wh, 128, f2b, h4, nullptr, NB, 128, 128, 1);
  k_gemm<<<12, 128, 0, stream>>>(h4, 128, f3wh, 128, f3b, nullptr, z, NB, DIM, 128, 0);

  // GMM tail
  k_head<<<1, NB, 0, stream>>>(z, cent, outp, dof, cls);
  k_lists<<<1, 32, 0, stream>>>(cls, idx, cnt, cwn, off, smap);
  k_build<<<(D2 * NBP) / 256, 256, 0, stream>>>(dof, smap, pairT, doT);
  k_p1<<<(NC * DIM + 255) / 256, 256, 0, stream>>>(dof, idx, cnt, cwn, gm1, mw, outp + 2560);
  k_p2<<<(NC * D2) / 256, 256, 0, stream>>>(dof, idx, cnt, cwn, gm2, mw, outp + 2561);
  k_p3<<<108, 128, 0, stream>>>(pairT, doT, cnt, cwn, off, gm3, mw, outp + 2562);
  k_p4<<<5184, 128, 0, stream>>>(pairT, cnt, cwn, off, gm4, mw, outp + 2563);
}
